// ConditionalARGDenoising_14250701488266
// MI455X (gfx1250) — compile-verified, matrix operand order checked by reading, not run on hardware
//
#include <hip/hip_runtime.h>
#include <hip/hip_bf16.h>

typedef __bf16 bf16;
typedef __attribute__((ext_vector_type(16))) __bf16 v16bf;
typedef __attribute__((ext_vector_type(8)))  __bf16 v8bf;
typedef __attribute__((ext_vector_type(8)))  float  v8f;

#define HID 128

__device__ __forceinline__ float silu_f(float v) { return v / (1.0f + __expf(-v)); }

// ---------------------------------------------------------------------------
// WMMA GEMM: Y[M,128] = act(A[M,Kpad] @ W + bias)   (bf16 in, f32 accum)
// A  : bf16 row-major [M, lda]  (lda == Kpad, multiple of 32)
// Wt : bf16 [128, lda]  (row n = column n of W, zero-padded to Kpad)
// outF (f32, optional; accumulate adds) / outH (bf16, optional), both [M,128]
// Block = 256 threads = 8 waves covering 64(M) x 128(N):
//   wave w owns N-tile 16*w and 4 M-tiles. Per K=32 step: load ALL fragments
//   (1 B + 4 A, distinct registers) before issuing the 4 WMMAs so the
//   scheduler can overlap loads with matrix ops (partial loadcnt waits).
// act: 0=none 1=silu 2=relu
// ---------------------------------------------------------------------------
__global__ __launch_bounds__(256) void gemm_wmma_k(
    const bf16* __restrict__ A, int lda,
    const bf16* __restrict__ Wt,
    const float* __restrict__ bias,
    float* __restrict__ outF, bf16* __restrict__ outH,
    int M, int Kpad, int act, int accumulate)
{
  const int wave = threadIdx.x >> 5;
  const int lane = threadIdx.x & 31;
  const int m0 = blockIdx.x * 64;
  const int n0 = wave * 16;
  const int kh  = lane >> 4;   // half selector per ISA 16-bit A layout
  const int idx = lane & 15;
  union V16 { v16bf v; v8bf h[2]; };

  // 4 M-tile row pointers, clamped so tail-block loads stay in bounds
  const bf16* arow0;
  const bf16* arow1;
  const bf16* arow2;
  const bf16* arow3;
  {
    int mr0 = m0 +  0; if (mr0 + 16 > M) mr0 = M - 16;
    int mr1 = m0 + 16; if (mr1 + 16 > M) mr1 = M - 16;
    int mr2 = m0 + 32; if (mr2 + 16 > M) mr2 = M - 16;
    int mr3 = m0 + 48; if (mr3 + 16 > M) mr3 = M - 16;
    arow0 = A + (size_t)(mr0 + idx) * lda + kh * 8;
    arow1 = A + (size_t)(mr1 + idx) * lda + kh * 8;
    arow2 = A + (size_t)(mr2 + idx) * lda + kh * 8;
    arow3 = A + (size_t)(mr3 + idx) * lda + kh * 8;
  }
  const bf16* brow = Wt + (size_t)(n0 + idx) * lda + kh * 8;

  v8f c0 = {}, c1 = {}, c2 = {}, c3 = {};
  for (int k = 0; k < Kpad; k += 32) {
    // ---- load phase: 10 independent b128 loads into distinct registers
    V16 b, a0, a1, a2, a3;
    b.h[0]  = *(const v8bf*)(brow  + k);       // K k+0..7   (lanes>=16: +8)
    b.h[1]  = *(const v8bf*)(brow  + k + 16);  // K k+16..23 (lanes>=16: +24)
    a0.h[0] = *(const v8bf*)(arow0 + k);
    a0.h[1] = *(const v8bf*)(arow0 + k + 16);
    a1.h[0] = *(const v8bf*)(arow1 + k);
    a1.h[1] = *(const v8bf*)(arow1 + k + 16);
    a2.h[0] = *(const v8bf*)(arow2 + k);
    a2.h[1] = *(const v8bf*)(arow2 + k + 16);
    a3.h[0] = *(const v8bf*)(arow3 + k);
    a3.h[1] = *(const v8bf*)(arow3 + k + 16);
    // ---- compute phase: one B fragment feeds 4 WMMAs
    c0 = __builtin_amdgcn_wmma_f32_16x16x32_bf16(false, a0.v, false, b.v,
                                                 (short)0, c0, false, false);
    c1 = __builtin_amdgcn_wmma_f32_16x16x32_bf16(false, a1.v, false, b.v,
                                                 (short)0, c1, false, false);
    c2 = __builtin_amdgcn_wmma_f32_16x16x32_bf16(false, a2.v, false, b.v,
                                                 (short)0, c2, false, false);
    c3 = __builtin_amdgcn_wmma_f32_16x16x32_bf16(false, a3.v, false, b.v,
                                                 (short)0, c3, false, false);
  }

  const int colg = n0 + idx;
  const float bv = bias ? bias[colg] : 0.0f;
  v8f acc[4] = {c0, c1, c2, c3};
#pragma unroll
  for (int mi = 0; mi < 4; ++mi) {
    const int mbase = m0 + mi * 16;
    if (mbase + 16 > M) continue;             // clamped duplicate tile: skip store
#pragma unroll
    for (int r = 0; r < 8; ++r) {
      int row = mbase + r + kh * 8;           // f32 C/D layout
      float v = acc[mi][r] + bv;
      if (act == 1)      v = silu_f(v);
      else if (act == 2) v = fmaxf(v, 0.0f);
      size_t o = (size_t)row * HID + colg;
      if (outF) outF[o] = accumulate ? (outF[o] + v) : v;
      if (outH) outH[o] = (bf16)v;
    }
  }
}

// W f32 [K,N=128] -> Wt bf16 [128, Kpad] (transposed, zero-padded)
__global__ void wt_prep_k(const float* __restrict__ W, bf16* __restrict__ Wt,
                          int K, int Kpad)
{
  int i = blockIdx.x * blockDim.x + threadIdx.x;
  if (i >= HID * Kpad) return;
  int n = i / Kpad, k = i % Kpad;
  Wt[i] = (k < K) ? (bf16)W[(size_t)k * HID + n] : (bf16)0.0f;
}

// f32 [rows, icols] -> bf16 [rows, ocols] zero-padded
__global__ void cvt_pad_k(const float* __restrict__ in, bf16* __restrict__ out,
                          int rows, int icols, int ocols)
{
  size_t i = (size_t)blockIdx.x * blockDim.x + threadIdx.x;
  if (i >= (size_t)rows * ocols) return;
  int r = (int)(i / ocols), c = (int)(i % ocols);
  out[i] = (c < icols) ? (bf16)in[(size_t)r * icols + c] : (bf16)0.0f;
}

// coord_diff + radial (+ optional normalize)
__global__ void edge_coord_k(const int* __restrict__ rows, const int* __restrict__ cols,
                             const float* __restrict__ coord,
                             float* __restrict__ cdiff, float* __restrict__ radial,
                             int E, int normalize)
{
  int e = blockIdx.x * blockDim.x + threadIdx.x;
  if (e >= E) return;
  int r = rows[e], c = cols[e];
  float dx = coord[r * 3 + 0] - coord[c * 3 + 0];
  float dy = coord[r * 3 + 1] - coord[c * 3 + 1];
  float dz = coord[r * 3 + 2] - coord[c * 3 + 2];
  float rad = dx * dx + dy * dy + dz * dz;
  radial[e] = rad;
  if (normalize) {
    float inv = 1.0f / (sqrtf(rad) + 1e-8f);
    dx *= inv; dy *= inv; dz *= inv;
  }
  cdiff[e * 3 + 0] = dx; cdiff[e * 3 + 1] = dy; cdiff[e * 3 + 2] = dz;
}

// e_in[e] = [h[row] (128) | h[col] (128) | radial (1) | attr (attr_w) | pad]
__global__ void build_ein_k(const int* __restrict__ rows, const int* __restrict__ cols,
                            const float* __restrict__ h, const float* __restrict__ radial,
                            const float* __restrict__ attr, int attr_w,
                            bf16* __restrict__ ein, int ldein)
{
  const int e = blockIdx.x;
  const int j = threadIdx.x;                  // 128 threads
  bf16* o = ein + (size_t)e * ldein;
  const int r = rows[e], c = cols[e];
  o[j]       = (bf16)h[(size_t)r * HID + j];
  o[HID + j] = (bf16)h[(size_t)c * HID + j];
  if (j == 0) o[256] = (bf16)radial[e];
  if (attr_w == HID)      o[257 + j] = (bf16)attr[(size_t)e * HID + j];
  else if (j == 0)        o[257]     = (bf16)attr[e];
  for (int t = 257 + attr_w + j; t < ldein; t += HID) o[t] = (bf16)0.0f;
}

// t[e] = dot(c1[e], w2)   (c1 bf16 [E,128], w2 f32 [128])
__global__ void coord_dot_k(const bf16* __restrict__ c1, const float* __restrict__ w2,
                            float* __restrict__ t, int E)
{
  int e = blockIdx.x * blockDim.x + threadIdx.x;
  if (e >= E) return;
  const bf16* p = c1 + (size_t)e * HID;
  float s = 0.0f;
  for (int j = 0; j < HID; ++j) s += (float)p[j] * w2[j];
  t[e] = s;
}

// segment sums via atomics: agg[row]+=m, aggc[row]+=cdiff*t, cnt[row]+=1
__global__ void scatter_k(const int* __restrict__ rows, const bf16* __restrict__ m,
                          const float* __restrict__ cdiff, const float* __restrict__ t,
                          float* __restrict__ agg, float* __restrict__ aggc,
                          float* __restrict__ cnt)
{
  const int e = blockIdx.x;
  const int j = threadIdx.x;                  // 128 threads
  const int r = rows[e];
  atomicAdd(&agg[(size_t)r * HID + j], (float)m[(size_t)e * HID + j]);
  if (j < 3)  atomicAdd(&aggc[r * 3 + j], cdiff[e * 3 + j] * t[e]);
  if (j == 3) atomicAdd(&cnt[r], 1.0f);
}

__global__ void coord_update_k(float* __restrict__ coord, const float* __restrict__ aggc,
                               const float* __restrict__ cnt, int N)
{
  int i = blockIdx.x * blockDim.x + threadIdx.x;
  if (i >= N * 3) return;
  coord[i] += aggc[i] / fmaxf(cnt[i / 3], 1.0f);
}

// nodecat[n] = [h[n] | agg[n]] as bf16 [N,256]
__global__ void build_nodecat_k(const float* __restrict__ h, const float* __restrict__ agg,
                                bf16* __restrict__ out)
{
  const int n = blockIdx.x;
  const int j = threadIdx.x;                  // 128 threads
  out[(size_t)n * 256 + j]       = (bf16)h[(size_t)n * HID + j];
  out[(size_t)n * 256 + HID + j] = (bf16)agg[(size_t)n * HID + j];
}

// h_e[e,j] = ea[e]*w[j] + b[j]
__global__ void edge_emb_k(const float* __restrict__ ea, const float* __restrict__ w,
                           const float* __restrict__ b, float* __restrict__ he, int E)
{
  size_t i = (size_t)blockIdx.x * blockDim.x + threadIdx.x;
  if (i >= (size_t)E * HID) return;
  int e = (int)(i >> 7), j = (int)(i & 127);
  he[i] = ea[e] * w[j] + b[j];
}

// g[col] = mean over rows of h[:,col]   (one block per column)
__global__ void mean_pool_k(const float* __restrict__ h, float* __restrict__ g, int N)
{
  __shared__ float sh[256];
  const int j = blockIdx.x;
  float s = 0.0f;
  for (int n = threadIdx.x; n < N; n += blockDim.x) s += h[(size_t)n * HID + j];
  sh[threadIdx.x] = s; __syncthreads();
  for (int w = blockDim.x / 2; w > 0; w >>= 1) {
    if (threadIdx.x < w) sh[threadIdx.x] += sh[threadIdx.x + w];
    __syncthreads();
  }
  if (threadIdx.x == 0) g[j] = sh[0] / (float)N;
}

// embed[j] = dot(g, fc_w[:,j]) + fc_b[j], j in [0,1280)
__global__ void fc_embed_k(const float* __restrict__ g, const float* __restrict__ W,
                           const float* __restrict__ b, float* __restrict__ embed)
{
  int j = blockIdx.x * blockDim.x + threadIdx.x;
  if (j >= 1280) return;
  float s = b[j];
  for (int k = 0; k < HID; ++k) s += g[k] * W[(size_t)k * 1280 + j];
  embed[j] = s;
}

// FiLM: h = scale*h + shift (per feature)
__global__ void film_k(float* __restrict__ h, const float* __restrict__ emb, int N)
{
  size_t i = (size_t)blockIdx.x * blockDim.x + threadIdx.x;
  if (i >= (size_t)N * HID) return;
  int j = (int)(i & 127);
  h[i] = emb[j] * h[i] + emb[HID + j];
}

// prediction head for the single needed node (row N-1)
__global__ void pred_final_k(const float* __restrict__ ge, const float* __restrict__ hv_last,
                             const float* __restrict__ w1, const float* __restrict__ b1,
                             const float* __restrict__ w2, const float* __restrict__ b2,
                             const float* __restrict__ w3, const float* __restrict__ b3,
                             float* __restrict__ out)
{
  __shared__ float z0[256], z1[HID], z2[HID];
  const int j = threadIdx.x;                  // 128 threads
  z0[j] = ge[j]; z0[HID + j] = hv_last[j];
  __syncthreads();
  float s = b1[j];
  for (int k = 0; k < 256; ++k) s += z0[k] * w1[(size_t)k * HID + j];
  z1[j] = fmaxf(s, 0.0f); __syncthreads();
  s = b2[j];
  for (int k = 0; k < HID; ++k) s += z1[k] * w2[(size_t)k * HID + j];
  z2[j] = fmaxf(s, 0.0f); __syncthreads();
  if (j < 112) {
    s = b3[j];
    for (int k = 0; k < HID; ++k) s += z2[k] * w3[(size_t)k * 112 + j];
    out[j] = s;
  }
}

// ---------------------------------------------------------------------------
// Host side
// ---------------------------------------------------------------------------
struct EgclW {
  const float *cb1, *cw1, *cw2, *eb1, *eb2, *ew1, *ew2, *nb1, *nb2, *nw1, *nw2;
};

struct Scratch {
  float *hv, *h, *he, *xc, *xv, *radial, *cdiff, *t;
  bf16 *ein, *m1, *mm, *nodecat, *xbf, *condbf, *hbf;
  float *agg, *aggc, *cnt, *g, *embed, *ge;
  bf16 *ew1t, *ew2t, *cw1t, *nw1t, *nw2t, *misc_t;
};

static inline void launch_gemm(const bf16* A, int lda, const bf16* Wt, const float* bias,
                               float* outF, bf16* outH, int M, int Kpad, int act, int acc,
                               hipStream_t s)
{
  gemm_wmma_k<<<dim3((M + 63) / 64), dim3(256), 0, s>>>(A, lda, Wt, bias, outF, outH,
                                                        M, Kpad, act, acc);
}

static inline void launch_wt(const float* W, bf16* Wt, int K, int Kpad, hipStream_t s)
{
  int n = HID * Kpad;
  wt_prep_k<<<dim3((n + 255) / 256), dim3(256), 0, s>>>(W, Wt, K, Kpad);
}

static void run_egcl(const EgclW& W, float* h, float* coord,
                     const float* attr, int attr_w,
                     const int* rows, const int* cols, int N, int E,
                     int normalize, Scratch& S, hipStream_t s)
{
  const int einK  = 257 + attr_w;                 // 258 or 385
  const int Kpad  = ((einK + 31) / 32) * 32;      // 288 or 416

  edge_coord_k<<<dim3((E + 255) / 256), dim3(256), 0, s>>>(rows, cols, coord, S.cdiff,
                                                           S.radial, E, normalize);
  build_ein_k<<<dim3(E), dim3(HID), 0, s>>>(rows, cols, h, S.radial, attr, attr_w,
                                            S.ein, Kpad);
  // edge MLP: m1 = silu(e_in @ W1 + b1); m = silu(m1 @ W2 + b2)
  launch_wt(W.ew1, S.ew1t, einK, Kpad, s);
  launch_gemm(S.ein, Kpad, S.ew1t, W.eb1, nullptr, S.m1, E, Kpad, 1, 0, s);
  launch_wt(W.ew2, S.ew2t, HID, HID, s);
  launch_gemm(S.m1, HID, S.ew2t, W.eb2, nullptr, S.mm, E, HID, 1, 0, s);
  // coord path: t = (silu(m @ Wc1 + bc1)) @ Wc2   (c1 reuses m1 buffer)
  launch_wt(W.cw1, S.cw1t, HID, HID, s);
  launch_gemm(S.mm, HID, S.cw1t, W.cb1, nullptr, S.m1, E, HID, 1, 0, s);
  coord_dot_k<<<dim3((E + 127) / 128), dim3(128), 0, s>>>(S.m1, W.cw2, S.t, E);
  // segment sums
  hipMemsetAsync(S.agg, 0, (size_t)N * HID * sizeof(float), s);
  hipMemsetAsync(S.aggc, 0, (size_t)N * 3 * sizeof(float), s);
  hipMemsetAsync(S.cnt, 0, (size_t)N * sizeof(float), s);
  scatter_k<<<dim3(E), dim3(HID), 0, s>>>(rows, S.mm, S.cdiff, S.t, S.agg, S.aggc, S.cnt);
  coord_update_k<<<dim3((N * 3 + 255) / 256), dim3(256), 0, s>>>(coord, S.aggc, S.cnt, N);
  // node MLP: h += (silu([h|agg] @ Wn1 + bn1)) @ Wn2 + bn2
  build_nodecat_k<<<dim3(N), dim3(HID), 0, s>>>(h, S.agg, S.nodecat);
  launch_wt(W.nw1, S.nw1t, 256, 256, s);
  launch_gemm(S.nodecat, 256, S.nw1t, W.nb1, nullptr, S.m1, N, 256, 1, 0, s);
  launch_wt(W.nw2, S.nw2t, HID, HID, s);
  launch_gemm(S.m1, HID, S.nw2t, W.nb2, h, nullptr, N, HID, 0, 1, s);
}

static EgclW load_egcl(void* const* d_in, int base)
{
  // alphabetical leaf order within each layer dict:
  // coord_b1, coord_w1, coord_w2, edge_b1, edge_b2, edge_w1, edge_w2,
  // node_b1, node_b2, node_w1, node_w2
  EgclW w;
  w.cb1 = (const float*)d_in[base + 0];
  w.cw1 = (const float*)d_in[base + 1];
  w.cw2 = (const float*)d_in[base + 2];
  w.eb1 = (const float*)d_in[base + 3];
  w.eb2 = (const float*)d_in[base + 4];
  w.ew1 = (const float*)d_in[base + 5];
  w.ew2 = (const float*)d_in[base + 6];
  w.nb1 = (const float*)d_in[base + 7];
  w.nb2 = (const float*)d_in[base + 8];
  w.nw1 = (const float*)d_in[base + 9];
  w.nw2 = (const float*)d_in[base + 10];
  return w;
}

extern "C" void kernel_launch(void* const* d_in, const int* in_sizes, int n_in,
                              void* d_out, int out_size, void* d_ws, size_t ws_size,
                              hipStream_t stream)
{
  // ------- inputs (jax tree-flatten order: dict keys sorted, lists in order)
  const float* x          = (const float*)d_in[0];   // [N,7,16]
  const int*   edge_index = (const int*)  d_in[1];   // [2,E]
  const float* edge_attr  = (const float*)d_in[2];   // [E]
  const float* x_coord    = (const float*)d_in[3];   // [N,3]
  const float* cond       = (const float*)d_in[4];   // [N,4,2]
  const int E = in_sizes[1] / 2;
  const int N = in_sizes[3] / 3;
  const int* rows = edge_index;
  const int* cols = edge_index + E;

  // params, alphabetical: edge_emb_b, edge_emb_w, enc_in_b, enc_in_w,
  // enc_layers(3x11), enc_out_b, enc_out_w, fc_b, fc_w, layers(5x11),
  // node_emb_b, node_emb_w, pred_b1, pred_b2, pred_b3, pred_w1, pred_w2, pred_w3
  const float* edge_emb_b = (const float*)d_in[5];
  const float* edge_emb_w = (const float*)d_in[6];
  const float* enc_in_b   = (const float*)d_in[7];
  const float* enc_in_w   = (const float*)d_in[8];
  const int enc_base = 9;                       // 3 layers x 11 leaves
  const float* enc_out_b  = (const float*)d_in[42];
  const float* enc_out_w  = (const float*)d_in[43];
  const float* fc_b       = (const float*)d_in[44];
  const float* fc_w       = (const float*)d_in[45];
  const int lay_base = 46;                      // 5 layers x 11 leaves
  const float* node_emb_b = (const float*)d_in[101];
  const float* node_emb_w = (const float*)d_in[102];
  const float* pred_b1    = (const float*)d_in[103];
  const float* pred_b2    = (const float*)d_in[104];
  const float* pred_b3    = (const float*)d_in[105];
  const float* pred_w1    = (const float*)d_in[106];
  const float* pred_w2    = (const float*)d_in[107];
  const float* pred_w3    = (const float*)d_in[108];
  (void)n_in; (void)ws_size;

  // ------- workspace carve-up
  char* p = (char*)d_ws;
  auto alloc = [&](size_t bytes) -> void* {
    void* r = (void*)p;
    p += (bytes + 255) & ~(size_t)255;
    return r;
  };
  Scratch S;
  S.hv      = (float*)alloc((size_t)N * HID * 4);
  S.h       = (float*)alloc((size_t)N * HID * 4);
  S.he      = (float*)alloc((size_t)E * HID * 4);
  S.xc      = (float*)alloc((size_t)N * 3 * 4);
  S.xv      = (float*)alloc((size_t)N * 3 * 4);
  S.radial  = (float*)alloc((size_t)E * 4);
  S.cdiff   = (float*)alloc((size_t)E * 3 * 4);
  S.t       = (float*)alloc((size_t)E * 4);
  S.ein     = (bf16*)alloc((size_t)E * 416 * 2);
  S.m1      = (bf16*)alloc((size_t)E * HID * 2);
  S.mm      = (bf16*)alloc((size_t)E * HID * 2);
  S.nodecat = (bf16*)alloc((size_t)N * 256 * 2);
  S.xbf     = (bf16*)alloc((size_t)N * HID * 2);
  S.condbf  = (bf16*)alloc((size_t)N * 32 * 2);
  S.hbf     = (bf16*)alloc((size_t)N * HID * 2);
  S.agg     = (float*)alloc((size_t)N * HID * 4);
  S.aggc    = (float*)alloc((size_t)N * 3 * 4);
  S.cnt     = (float*)alloc((size_t)N * 4);
  S.g       = (float*)alloc(HID * 4);
  S.embed   = (float*)alloc(1280 * 4);
  S.ge      = (float*)alloc(HID * 4);
  S.ew1t    = (bf16*)alloc((size_t)HID * 416 * 2);
  S.ew2t    = (bf16*)alloc((size_t)HID * HID * 2);
  S.cw1t    = (bf16*)alloc((size_t)HID * HID * 2);
  S.nw1t    = (bf16*)alloc((size_t)HID * 256 * 2);
  S.nw2t    = (bf16*)alloc((size_t)HID * HID * 2);
  S.misc_t  = (bf16*)alloc((size_t)HID * HID * 2);

  // ------- coordinate copies
  hipMemcpyAsync(S.xc, x_coord, (size_t)N * 3 * 4, hipMemcpyDeviceToDevice, stream);
  hipMemcpyAsync(S.xv, x_coord, (size_t)N * 3 * 4, hipMemcpyDeviceToDevice, stream);

  // ------- embeddings
  // h_v = x.reshape(N,112) @ node_emb_w + b     (pad K 112 -> 128)
  cvt_pad_k<<<dim3(((size_t)N * HID + 255) / 256), dim3(256), 0, stream>>>(x, S.xbf, N, 112, HID);
  launch_wt(node_emb_w, S.misc_t, 112, HID, stream);
  launch_gemm(S.xbf, HID, S.misc_t, node_emb_b, S.hv, nullptr, N, HID, 0, 0, stream);
  // h_e = ea @ edge_emb_w + b (outer product)
  edge_emb_k<<<dim3(((size_t)E * HID + 255) / 256), dim3(256), 0, stream>>>(edge_attr, edge_emb_w,
                                                                            edge_emb_b, S.he, E);
  // h = cond.reshape(N,8) @ enc_in_w + b        (pad K 8 -> 32)
  cvt_pad_k<<<dim3(((size_t)N * 32 + 255) / 256), dim3(256), 0, stream>>>(cond, S.condbf, N, 8, 32);
  launch_wt(enc_in_w, S.misc_t, 8, 32, stream);
  launch_gemm(S.condbf, 32, S.misc_t, enc_in_b, S.h, nullptr, N, 32, 0, 0, stream);

  // ------- conditioning encoder: 3 EGCL (normalize=False, attr = ea width 1)
  for (int l = 0; l < 3; ++l) {
    EgclW w = load_egcl(d_in, enc_base + 11 * l);
    run_egcl(w, S.h, S.xc, edge_attr, 1, rows, cols, N, E, /*normalize=*/0, S, stream);
  }
  // h = h @ enc_out_w + b ; g = mean(h) ; embed = g @ fc_w + fc_b
  cvt_pad_k<<<dim3(((size_t)N * HID + 255) / 256), dim3(256), 0, stream>>>(S.h, S.hbf, N, HID, HID);
  launch_wt(enc_out_w, S.misc_t, HID, HID, stream);
  launch_gemm(S.hbf, HID, S.misc_t, enc_out_b, S.h, nullptr, N, HID, 0, 0, stream);
  mean_pool_k<<<dim3(HID), dim3(256), 0, stream>>>(S.h, S.g, N);
  fc_embed_k<<<dim3((1280 + 255) / 256), dim3(256), 0, stream>>>(S.g, fc_w, fc_b, S.embed);

  // ------- FiLM-conditioned denoising stack: 5 EGCL (normalize=True, attr = h_e)
  for (int l = 0; l < 5; ++l) {
    film_k<<<dim3(((size_t)N * HID + 255) / 256), dim3(256), 0, stream>>>(S.hv,
                                                                          S.embed + l * 256, N);
    EgclW w = load_egcl(d_in, lay_base + 11 * l);
    run_egcl(w, S.hv, S.xv, S.he, HID, rows, cols, N, E, /*normalize=*/1, S, stream);
  }

  // ------- prediction head (only node N-1 is needed) + outputs
  mean_pool_k<<<dim3(HID), dim3(256), 0, stream>>>(S.hv, S.ge, N);
  float* out = (float*)d_out;
  pred_final_k<<<dim3(1), dim3(HID), 0, stream>>>(S.ge, S.hv + (size_t)(N - 1) * HID,
                                                  pred_w1, pred_b1, pred_w2, pred_b2,
                                                  pred_w3, pred_b3, out);
  hipMemcpyAsync(out + 112, S.xv, (size_t)N * 3 * 4, hipMemcpyDeviceToDevice, stream);
  (void)out_size;
}
